// RNNCell_26551487824698
// MI455X (gfx1250) — matrix operand order repaired and checked by reading, not compiled, not run on hardware
//
#include <hip/hip_runtime.h>
#include <hip/hip_bf16.h>
#include <stdint.h>

// LSTM cell fused kernel for MI455X (gfx1250), bf16 WMMA with f32 accumulate.
// B=65536, I=H=256, gate size 4H=1024.
// Each block: 256 threads (8 waves), covers 256 batch rows x 16 hidden cols.
// Each wave: two 16-row M tiles -> every LDS B-fragment feeds 2 WMMAs.

typedef __bf16 v16bf __attribute__((ext_vector_type(16)));
typedef float  v8f   __attribute__((ext_vector_type(8)));

#define BATCH 65536
#define HDIM  256
#define IDIM  256

__device__ __forceinline__ float fast_sigmoid(float x) {
    return 1.0f / (1.0f + __expf(-x));
}
__device__ __forceinline__ float fast_tanh(float x) {
    float e = __expf(2.0f * x);
    return (e - 1.0f) / (e + 1.0f);
}

__device__ __forceinline__ uint32_t pack_bf16x2(float lo, float hi) {
    __bf16 l = (__bf16)lo, h = (__bf16)hi;
    uint16_t lu, hu;
    __builtin_memcpy(&lu, &l, 2);
    __builtin_memcpy(&hu, &h, 2);
    return (uint32_t)lu | ((uint32_t)hu << 16);
}

// A-fragment (16x32 bf16, row-major f32 source). Lane
// (half=lane>>4, row=lane&15) holds 16 contiguous K values:
// [k0+16*half, +16) -> elements [0..15] (matches B-fragment order).
__device__ __forceinline__ v16bf load_a_frag(const float* __restrict__ rowp,
                                             int k0, int half) {
    const float4* p = (const float4*)(rowp + k0 + 16 * half);
    float4 f0 = p[0], f1 = p[1];
    const float4* q = (const float4*)(rowp + k0 + 16 * half + 8);
    float4 f2 = q[0], f3 = q[1];
    v16bf a;
    a[0]  = (__bf16)f0.x; a[1]  = (__bf16)f0.y; a[2]  = (__bf16)f0.z; a[3]  = (__bf16)f0.w;
    a[4]  = (__bf16)f1.x; a[5]  = (__bf16)f1.y; a[6]  = (__bf16)f1.z; a[7]  = (__bf16)f1.w;
    a[8]  = (__bf16)f2.x; a[9]  = (__bf16)f2.y; a[10] = (__bf16)f2.z; a[11] = (__bf16)f2.w;
    a[12] = (__bf16)f3.x; a[13] = (__bf16)f3.y; a[14] = (__bf16)f3.z; a[15] = (__bf16)f3.w;
    return a;
}

// B-fragment (32x16 bf16): lane holds column N=lane&15, 16 contiguous K values
// starting at k0+16*(lane>>4). LDS stores weights K-contiguous per (mat,gate,col).
__device__ __forceinline__ v16bf load_b_frag(const uint32_t* ldsw, int mat, int g,
                                             int col, int kbase) {
    int idx = ((((mat << 2) + g) << 4) + col) * 128 + (kbase >> 1); // uint32 pairs
    return *(const v16bf*)(ldsw + idx);
}

__launch_bounds__(256)
__global__ void lstm_cell_wmma_bf16(const float* __restrict__ x,
                                    const float* __restrict__ h,
                                    const float* __restrict__ c,
                                    const float* __restrict__ w_ih,
                                    const float* __restrict__ w_hh,
                                    const float* __restrict__ b_ih,
                                    const float* __restrict__ b_hh,
                                    float* __restrict__ out) {
    // 2 mats * 4 gates * 16 cols * 256 K * bf16 = 64 KB
    __shared__ __align__(32) uint32_t ldsw[16384];

    const int tid = threadIdx.x;
    const int n0  = blockIdx.y << 4;   // hidden-column block

    // ---- Stage bf16 weight slice into LDS (all 8 waves cooperate) ----
    // Flat uint32-pair index e: [mat(1)][gate(2)][col(4)][kpair(7)]
    #pragma unroll 4
    for (int j = 0; j < 64; ++j) {
        int e    = tid + (j << 8);
        int kp   = e & 127;
        int col  = (e >> 7) & 15;
        int gate = (e >> 11) & 3;
        int mat  = e >> 13;
        const float* src = (mat ? w_hh : w_ih)
                         + ((gate << 8) + n0 + col) * 256 + (kp << 1);
        float2 f = *(const float2*)src;
        ldsw[e] = pack_bf16x2(f.x, f.y);
    }
    __syncthreads();

    const int lane = tid & 31;
    const int wave = tid >> 5;
    const int half = lane >> 4;
    const int lcol = lane & 15;                         // N within tile
    const long m0  = (long)blockIdx.x * 256 + wave * 32; // first of 2 M tiles

    const float* xrow0 = x + (m0 + lcol) * (long)IDIM;       // tile 0 A rows
    const float* hrow0 = h + (m0 + lcol) * (long)HDIM;
    const float* xrow1 = x + (m0 + 16 + lcol) * (long)IDIM;  // tile 1 A rows
    const float* hrow1 = h + (m0 + 16 + lcol) * (long)HDIM;

    // ---- Seed accumulators with biases (bias depends only on N) ----
    v8f acc0[4], acc1[4];
    #pragma unroll
    for (int g = 0; g < 4; ++g) {
        float bb = b_ih[(g << 8) + n0 + lcol] + b_hh[(g << 8) + n0 + lcol];
        #pragma unroll
        for (int r = 0; r < 8; ++r) { acc0[g][r] = bb; acc1[g][r] = bb; }
    }

    // ---- K loop: 8 chunks of 32; 16 wmma per chunk (4 gates x {x,h} x 2 M) ----
    #pragma unroll
    for (int k0 = 0; k0 < 256; k0 += 32) {
        if (k0 + 32 < 256) {
            __builtin_prefetch(xrow0 + k0 + 32, 0, 1);   // global_prefetch_b8
            __builtin_prefetch(hrow0 + k0 + 32, 0, 1);
            __builtin_prefetch(xrow1 + k0 + 32, 0, 1);
            __builtin_prefetch(hrow1 + k0 + 32, 0, 1);
        }
        v16bf ax0 = load_a_frag(xrow0, k0, half);
        v16bf ax1 = load_a_frag(xrow1, k0, half);
        v16bf ah0 = load_a_frag(hrow0, k0, half);
        v16bf ah1 = load_a_frag(hrow1, k0, half);
        const int kb = k0 + 16 * half;
        #pragma unroll
        for (int g = 0; g < 4; ++g) {
            v16bf bx = load_b_frag(ldsw, 0, g, lcol, kb);
            acc0[g] = __builtin_amdgcn_wmma_f32_16x16x32_bf16(
                          false, ax0, false, bx, (short)0, acc0[g], false, false);
            acc1[g] = __builtin_amdgcn_wmma_f32_16x16x32_bf16(
                          false, ax1, false, bx, (short)0, acc1[g], false, false);
            v16bf bh = load_b_frag(ldsw, 1, g, lcol, kb);
            acc0[g] = __builtin_amdgcn_wmma_f32_16x16x32_bf16(
                          false, ah0, false, bh, (short)0, acc0[g], false, false);
            acc1[g] = __builtin_amdgcn_wmma_f32_16x16x32_bf16(
                          false, ah1, false, bh, (short)0, acc1[g], false, false);
        }
    }

    // ---- Fused LSTM epilogue (C/D layout: VGPR r -> M=r+8*half, N=lcol) ----
    const int  outcol = n0 + lcol;
    const long cbase  = (long)BATCH * HDIM;
    #pragma unroll
    for (int r = 0; r < 8; ++r) {
        // tile 0
        {
            long row = m0 + r + 8 * half;
            float ig = fast_sigmoid(acc0[0][r]);
            float fg = fast_sigmoid(acc0[1][r]);
            float gg = fast_tanh(acc0[2][r]);
            float og = fast_sigmoid(acc0[3][r]);
            float co = c[row * HDIM + outcol];
            float cn = fg * co + ig * gg;
            float hn = og * fast_tanh(cn);
            out[row * HDIM + outcol]         = hn;   // h_new
            out[cbase + row * HDIM + outcol] = cn;   // c_new
        }
        // tile 1
        {
            long row = m0 + 16 + r + 8 * half;
            float ig = fast_sigmoid(acc1[0][r]);
            float fg = fast_sigmoid(acc1[1][r]);
            float gg = fast_tanh(acc1[2][r]);
            float og = fast_sigmoid(acc1[3][r]);
            float co = c[row * HDIM + outcol];
            float cn = fg * co + ig * gg;
            float hn = og * fast_tanh(cn);
            out[row * HDIM + outcol]         = hn;
            out[cbase + row * HDIM + outcol] = cn;
        }
    }
}

extern "C" void kernel_launch(void* const* d_in, const int* in_sizes, int n_in,
                              void* d_out, int out_size, void* d_ws, size_t ws_size,
                              hipStream_t stream) {
    (void)in_sizes; (void)n_in; (void)out_size; (void)d_ws; (void)ws_size;
    const float* x    = (const float*)d_in[0];
    const float* h    = (const float*)d_in[1];
    const float* c    = (const float*)d_in[2];
    const float* w_ih = (const float*)d_in[3];
    const float* w_hh = (const float*)d_in[4];
    const float* b_ih = (const float*)d_in[5];
    const float* b_hh = (const float*)d_in[6];
    dim3 grid(BATCH / 256, HDIM / 16);   // 256 x 16 blocks, 8 waves each
    lstm_cell_wmma_bf16<<<grid, 256, 0, stream>>>(
        x, h, c, w_ih, w_hh, b_ih, b_hh, (float*)d_out);
}
